// HBitLinear_3178275799206
// MI455X (gfx1250) — 8-bit matrix operand order repaired and checked by reading, compiled, not run on hardware
//
#include <hip/hip_runtime.h>
#include <stdint.h>

#define K_FEAT 1024

typedef int   v8i __attribute__((ext_vector_type(8)));
typedef int   v4i_t __attribute__((ext_vector_type(4)));
typedef float v4f_t __attribute__((ext_vector_type(4)));

__global__ __launch_bounds__(256) void wquant_kernel(const float* __restrict__ w,
                                                     int8_t* __restrict__ wq,
                                                     float* __restrict__ sw,
                                                     int nrows) {
    const int lane = threadIdx.x & 31;
    const int wave = threadIdx.x >> 5;
    const int row  = blockIdx.x * 8 + wave;
    if (row >= nrows) return;

    float v[32];
    const v4f_t* p = (const v4f_t*)(w + (size_t)row * K_FEAT + lane * 32);
#pragma unroll
    for (int j = 0; j < 8; ++j) {
        v4f_t t = p[j];
        v[4*j+0] = t.x; v[4*j+1] = t.y; v[4*j+2] = t.z; v[4*j+3] = t.w;
    }

    float s = 0.0f;
#pragma unroll
    for (int j = 0; j < 32; ++j) s += fabsf(v[j]);
#pragma unroll
    for (int m = 16; m >= 1; m >>= 1) s += __shfl_xor(s, m, 32);

    const float scale = fmaxf(s * (1.0f / (float)K_FEAT), 1e-5f);
    const float inv   = 1.0f / scale;

    int packed[8];
#pragma unroll
    for (int j = 0; j < 8; ++j) {
        int pk = 0;
#pragma unroll
        for (int b = 0; b < 4; ++b) {
            float n = v[4*j+b] * inv;
            int   t = (n > 0.5f) ? 1 : ((n < -0.5f) ? -1 : 0);
            pk |= (t & 0xFF) << (8 * b);
        }
        packed[j] = pk;
    }
    v4i_t* dst = (v4i_t*)(wq + (size_t)row * K_FEAT + lane * 32);
    dst[0] = (v4i_t){packed[0], packed[1], packed[2], packed[3]};
    dst[1] = (v4i_t){packed[4], packed[5], packed[6], packed[7]};
    if (lane == 0) sw[row] = scale;
}

__global__ __launch_bounds__(256) void fwht_quant_kernel(const float* __restrict__ x,
                                                         int8_t* __restrict__ xq,
                                                         float* __restrict__ sx,
                                                         int ntok) {
    const int lane = threadIdx.x & 31;
    const int wave = threadIdx.x >> 5;
    const int tok  = blockIdx.x * 8 + wave;
    if (tok >= ntok) return;

    float r[32];
    const v4f_t* p = (const v4f_t*)(x + (size_t)tok * K_FEAT + lane * 32);
#pragma unroll
    for (int j = 0; j < 8; ++j) {
        v4f_t t = __builtin_nontemporal_load(p + j);
        r[4*j+0] = t.x; r[4*j+1] = t.y; r[4*j+2] = t.z; r[4*j+3] = t.w;
    }

#pragma unroll
    for (int d = 1; d < 32; d <<= 1) {
#pragma unroll
        for (int i = 0; i < 32; ++i) {
            if ((i & d) == 0) {
                float a = r[i], b = r[i + d];
                r[i] = a + b;
                r[i + d] = a - b;
            }
        }
    }
#pragma unroll
    for (int m = 1; m < 32; m <<= 1) {
        const bool hi = (lane & m) != 0;
#pragma unroll
        for (int j = 0; j < 32; ++j) {
            float o = __shfl_xor(r[j], m, 32);
            r[j] = hi ? (o - r[j]) : (r[j] + o);
        }
    }

    float mx = 0.0f;
#pragma unroll
    for (int j = 0; j < 32; ++j) mx = fmaxf(mx, fabsf(r[j]));
#pragma unroll
    for (int m = 16; m >= 1; m >>= 1) mx = fmaxf(mx, __shfl_xor(mx, m, 32));

    const float scale = fmaxf(mx * (1.0f / 32.0f), 1e-5f) * (1.0f / 7.0f);
    const float f     = 1.0f / (32.0f * scale);

    int packed[8];
#pragma unroll
    for (int j = 0; j < 8; ++j) {
        int pk = 0;
#pragma unroll
        for (int b = 0; b < 4; ++b) {
            float q = rintf(r[4*j+b] * f);
            q = fminf(7.0f, fmaxf(-8.0f, q));
            int qi = (int)q;
            pk |= (qi & 0xFF) << (8 * b);
        }
        packed[j] = pk;
    }
    v4i_t* dst = (v4i_t*)(xq + (size_t)tok * K_FEAT + lane * 32);
    dst[0] = (v4i_t){packed[0], packed[1], packed[2], packed[3]};
    dst[1] = (v4i_t){packed[4], packed[5], packed[6], packed[7]};
    if (lane == 0) sx[tok] = scale;
}

__device__ __forceinline__ void load_a_frags(const int8_t* __restrict__ xq,
                                             int m0, int l16, int half, int k0,
                                             v8i a[2]) {
#pragma unroll
    for (int i = 0; i < 2; ++i) {
        const int* arow = (const int*)(xq + (size_t)(m0 + i * 16 + l16) * K_FEAT + k0);
        const int base = half * 4;
        a[i][0] = arow[base + 0];  a[i][1] = arow[base + 1];
        a[i][2] = arow[base + 2];  a[i][3] = arow[base + 3];
        a[i][4] = arow[base + 8];  a[i][5] = arow[base + 9];
        a[i][6] = arow[base + 10]; a[i][7] = arow[base + 11];
    }
}

__device__ __forceinline__ void load_b_frags(const int8_t* __restrict__ wq,
                                             int n0, int l16, int half, int k0,
                                             v8i b[4]) {
#pragma unroll
    for (int j = 0; j < 4; ++j) {
        const int* brow = (const int*)(wq + (size_t)(n0 + j * 16 + l16) * K_FEAT + k0);
        const int base = half * 4;
#pragma unroll
        for (int e = 0; e < 4; ++e) b[j][e]     = brow[base + e];
#pragma unroll
        for (int e = 0; e < 4; ++e) b[j][4 + e] = brow[base + 8 + e];
    }
}

__global__ __launch_bounds__(256) void gemm_iu8_kernel(const int8_t* __restrict__ xq,
                                                       const float* __restrict__ sx,
                                                       const int8_t* __restrict__ wq,
                                                       const float* __restrict__ sw,
                                                       float* __restrict__ out,
                                                       int M, int N) {
    const int lane = threadIdx.x & 31;
    const int wave = threadIdx.x >> 5;
    const int half = lane >> 4;
    const int l16  = lane & 15;
    const int wm   = wave & 3;
    const int wn   = wave >> 2;
    const int m0   = blockIdx.x * 128 + wm * 32;
    const int n0   = blockIdx.y * 128 + wn * 64;

    v8i acc[2][4];
#pragma unroll
    for (int i = 0; i < 2; ++i)
#pragma unroll
        for (int j = 0; j < 4; ++j)
#pragma unroll
            for (int e = 0; e < 8; ++e) acc[i][j][e] = 0;

    v8i a0[2], b0[4], a1[2], b1[4];
    load_a_frags(xq, m0, l16, half, 0, a0);
    load_b_frags(wq, n0, l16, half, 0, b0);

#pragma unroll 1
    for (int k0 = 0; k0 < K_FEAT; k0 += 128) {
        load_a_frags(xq, m0, l16, half, k0 + 64, a1);
        load_b_frags(wq, n0, l16, half, k0 + 64, b1);
#pragma unroll
        for (int i = 0; i < 2; ++i)
#pragma unroll
            for (int j = 0; j < 4; ++j)
                acc[i][j] = __builtin_amdgcn_wmma_i32_16x16x64_iu8(
                    true, a0[i], true, b0[j], acc[i][j], false, false);

        if (k0 + 128 < K_FEAT) {
            load_a_frags(xq, m0, l16, half, k0 + 128, a0);
            load_b_frags(wq, n0, l16, half, k0 + 128, b0);
        }
#pragma unroll
        for (int i = 0; i < 2; ++i)
#pragma unroll
            for (int j = 0; j < 4; ++j)
                acc[i][j] = __builtin_amdgcn_wmma_i32_16x16x64_iu8(
                    true, a1[i], true, b1[j], acc[i][j], false, false);
    }

    float sxv[2][8];
#pragma unroll
    for (int i = 0; i < 2; ++i)
#pragma unroll
        for (int rI = 0; rI < 8; ++rI)
            sxv[i][rI] = sx[m0 + i * 16 + half * 8 + rI];

#pragma unroll
    for (int j = 0; j < 4; ++j) {
        const int   n   = n0 + j * 16 + l16;
        const float swv = sw[n];
#pragma unroll
        for (int i = 0; i < 2; ++i) {
#pragma unroll
            for (int rI = 0; rI < 8; ++rI) {
                const int mrow = m0 + i * 16 + half * 8 + rI;
                __builtin_nontemporal_store((float)acc[i][j][rI] * sxv[i][rI] * swv,
                                            &out[(size_t)mrow * N + n]);
            }
        }
    }
}

extern "C" void kernel_launch(void* const* d_in, const int* in_sizes, int n_in,
                              void* d_out, int out_size, void* d_ws, size_t ws_size,
                              hipStream_t stream) {
    const float* x = (const float*)d_in[0];
    const float* w = (const float*)d_in[1];
    float* out = (float*)d_out;

    const int M = in_sizes[0] / K_FEAT;
    const int N = in_sizes[1] / K_FEAT;

    char* ws = (char*)d_ws;
    int8_t* xq = (int8_t*)ws;
    size_t off = (size_t)M * K_FEAT;
    float* sx = (float*)(ws + off); off += (size_t)M * sizeof(float);
    int8_t* wq = (int8_t*)(ws + off); off += (size_t)N * K_FEAT;
    float* sw = (float*)(ws + off);

    wquant_kernel<<<(N + 7) / 8, 256, 0, stream>>>(w, wq, sw, N);
    fwht_quant_kernel<<<(M + 7) / 8, 256, 0, stream>>>(x, xq, sx, M);
    gemm_iu8_kernel<<<dim3(M / 128, N / 128), 256, 0, stream>>>(xq, sx, wq, sw, out, M, N);
}
